// DrugChatCompoundEncoder_3736621548050
// MI455X (gfx1250) — matrix operand order repaired and checked by reading, not compiled, not run on hardware
//
#include <hip/hip_runtime.h>
#include <hip/hip_bf16.h>
#include <stdint.h>

#define N_NODES 131072
#define N_EDGES 262144
#define NGRAPH  4096
#define DIM     300
#define NLAYER  5
#define MAXN    96
#define D2      600
#define KPAD1   320   // K padding for GEMM1 (300 -> 320, 10 steps of 32)
#define NPAD1   640   // padded col count for W1^T (600 -> 640, 10 tiles of 64)
#define KPAD2   608   // K padding for GEMM2 (600 -> 608, 19 steps of 32)
#define NPAD2   320   // padded col count for W2^T (300 -> 320, 5 tiles of 64)
#define BN_EPS  1e-5f

typedef __bf16 bf16;
typedef bf16  v16bf __attribute__((ext_vector_type(16)));
typedef bf16  v8bf  __attribute__((ext_vector_type(8)));
typedef float v8f   __attribute__((ext_vector_type(8)));

__device__ __forceinline__ bf16 f2bf(float f) {
  union { float f; uint32_t u; } c; c.f = f;
  uint32_t r = (c.u + 0x7FFFu + ((c.u >> 16) & 1u)) >> 16;   // round-to-nearest-even
  union { uint16_t u; bf16 b; } o; o.u = (uint16_t)r;
  return o.b;
}

// ---------------------------------------------------------------------------
// h[i,:] = atom_emb1[x[i,0],:] + atom_emb2[x[i,1],:]
__global__ void k_init_h(const int* __restrict__ x,
                         const float* __restrict__ ae1,
                         const float* __restrict__ ae2,
                         float* __restrict__ h) {
  int i = blockIdx.x;
  int a0 = x[2 * i], a1 = x[2 * i + 1];
  for (int d = threadIdx.x; d < DIM; d += blockDim.x)
    h[(size_t)i * DIM + d] = ae1[(size_t)a0 * DIM + d] + ae2[(size_t)a1 * DIM + d];
}

// agg[i,:] = h[i,:] + edge_emb1[l][SELF_LOOP=4] + edge_emb2[l][0]   (self-loop init)
__global__ void k_init_agg_self(const float* __restrict__ h,
                                const float* __restrict__ ee1l,
                                const float* __restrict__ ee2l,
                                float* __restrict__ agg) {
  int i = blockIdx.x;
  for (int d = threadIdx.x; d < DIM; d += blockDim.x)
    agg[(size_t)i * DIM + d] = h[(size_t)i * DIM + d] + ee1l[4 * DIM + d] + ee2l[d];
}

// agg[dst,:] += h[src,:] + edge_emb1[l][bond] + edge_emb2[l][dir]
__global__ void k_edge_scatter(const float* __restrict__ h,
                               const int* __restrict__ edge_index,
                               const int* __restrict__ edge_attr,
                               const float* __restrict__ ee1l,
                               const float* __restrict__ ee2l,
                               float* __restrict__ agg) {
  int j = blockIdx.x;
  int src  = edge_index[j];
  int dst  = edge_index[N_EDGES + j];
  int bond = edge_attr[2 * j];
  int dir  = edge_attr[2 * j + 1];
  const float* e1 = ee1l + (size_t)bond * DIM;
  const float* e2 = ee2l + (size_t)dir * DIM;
  for (int d = threadIdx.x; d < DIM; d += blockDim.x) {
    float v = h[(size_t)src * DIM + d] + e1[d] + e2[d];
    atomicAdd(agg + (size_t)dst * DIM + d, v);
  }
}

// fp32 agg [N,300] -> bf16 A [N,320] (zero K-pad)
__global__ void k_conv_A(const float* __restrict__ agg, bf16* __restrict__ Abf) {
  int i = blockIdx.x;
  for (int k = threadIdx.x; k < KPAD1; k += blockDim.x)
    Abf[(size_t)i * KPAD1 + k] = (k < DIM) ? f2bf(agg[(size_t)i * DIM + k]) : f2bf(0.0f);
}

// W1[l] fp32 [300,600] -> bf16 W1T [640,320] (transposed, zero-padded)
__global__ void k_conv_W1T(const float* __restrict__ W1l, bf16* __restrict__ W1T) {
  int n = blockIdx.x;                       // 0..639
  for (int k = threadIdx.x; k < KPAD1; k += blockDim.x) {
    float v = (n < D2 && k < DIM) ? W1l[(size_t)k * D2 + n] : 0.0f;
    W1T[(size_t)n * KPAD1 + k] = f2bf(v);
  }
}

// W2[l] fp32 [600,300] -> bf16 W2T [320,608]
__global__ void k_conv_W2T(const float* __restrict__ W2l, bf16* __restrict__ W2T) {
  int n = blockIdx.x;                       // 0..319
  for (int k = threadIdx.x; k < KPAD2; k += blockDim.x) {
    float v = (n < DIM && k < D2) ? W2l[(size_t)k * DIM + n] : 0.0f;
    W2T[(size_t)n * KPAD2 + k] = f2bf(v);
  }
}

// ---------------------------------------------------------------------------
// LDS-staged WMMA GEMM: out = epilogue(A[M,Ka] x Bt[colsPad,Ka]^T + bias)
// Block tile 128x64, 8 waves, each wave 16 rows x four 16x16 f32 accumulators.
// MODE 0: bias + ReLU -> bf16 out (GEMM1).  MODE 1: bias + BN (+ReLU) -> f32 out.
template <int MODE>
__global__ __launch_bounds__(256)
void gin_gemm_wmma(const bf16* __restrict__ A,
                   const bf16* __restrict__ Bt,
                   const float* __restrict__ bias,
                   const float* __restrict__ gamma,
                   const float* __restrict__ beta,
                   const float* __restrict__ mean,
                   const float* __restrict__ var,
                   int relu, int Ka, int biasCols, int storeCols,
                   float* __restrict__ outF, bf16* __restrict__ outB,
                   int outStride) {
  __shared__ __align__(64) bf16 ldsA[128][32];
  __shared__ __align__(64) bf16 ldsB[64][32];

  const int t    = threadIdx.x;
  const int lane = t & 31;
  const int wave = t >> 5;
  const int m    = lane & 15;       // row/col within 16-subtile
  const int half = lane >> 4;       // K half-block selector (ISA A/B layout)
  const int rowBase = blockIdx.x * 128;
  const int colBase = blockIdx.y * 64;

  v8f acc0 = {}, acc1 = {}, acc2 = {}, acc3 = {};

  const int aRow  = t >> 1, aKoff = (t & 1) * 16;   // 16 halves (32B) per thread
  const int bCol  = t >> 2, bKoff = (t & 3) * 8;    // 8 halves (16B) per thread

  for (int k0 = 0; k0 < Ka; k0 += 32) {
    // --- stage A tile (128x32 bf16) ---
    {
      const uint4* src = (const uint4*)(A + (size_t)(rowBase + aRow) * Ka + k0 + aKoff);
      uint4 d0 = src[0];
      uint4 d1 = src[1];
      if (k0 + 32 < Ka)
        __builtin_prefetch(A + (size_t)(rowBase + aRow) * Ka + k0 + 32 + aKoff, 0, 1);
      uint4* dst = (uint4*)&ldsA[aRow][aKoff];
      dst[0] = d0; dst[1] = d1;
    }
    // --- stage B tile transposed (64 cols x 32 k, bf16) ---
    {
      const uint4* src = (const uint4*)(Bt + (size_t)(colBase + bCol) * Ka + k0 + bKoff);
      *(uint4*)&ldsB[bCol][bKoff] = *src;
    }
    __syncthreads();

    // A fragment: lane<16 -> K 0..15, lane>=16 -> K 16..31
    const v8bf* arow = (const v8bf*)&ldsA[wave * 16 + m][0];
    v8bf alo = arow[2 * half];
    v8bf ahi = arow[2 * half + 1];
    v16bf afrag = __builtin_shufflevector(alo, ahi,
        0, 1, 2, 3, 4, 5, 6, 7, 8, 9, 10, 11, 12, 13, 14, 15);

    // B fragments: lane<16 -> K 0..15, lane>=16 -> K 16..31 of column (lane%16)
    v16bf b0 = *(const v16bf*)&ldsB[ 0 + m][half * 16];
    v16bf b1 = *(const v16bf*)&ldsB[16 + m][half * 16];
    v16bf b2 = *(const v16bf*)&ldsB[32 + m][half * 16];
    v16bf b3 = *(const v16bf*)&ldsB[48 + m][half * 16];

    acc0 = __builtin_amdgcn_wmma_f32_16x16x32_bf16(false, afrag, false, b0, (short)0, acc0, false, false);
    acc1 = __builtin_amdgcn_wmma_f32_16x16x32_bf16(false, afrag, false, b1, (short)0, acc1, false, false);
    acc2 = __builtin_amdgcn_wmma_f32_16x16x32_bf16(false, afrag, false, b2, (short)0, acc2, false, false);
    acc3 = __builtin_amdgcn_wmma_f32_16x16x32_bf16(false, afrag, false, b3, (short)0, acc3, false, false);
    __syncthreads();
  }

  v8f accs[4] = {acc0, acc1, acc2, acc3};
  #pragma unroll
  for (int j = 0; j < 4; ++j) {
    int col = colBase + j * 16 + m;
    if (col >= storeCols) continue;
    float bv = (col < biasCols) ? bias[col] : 0.0f;
    float g = 0.f, be = 0.f, mu = 0.f, iv = 0.f;
    if (MODE == 1) {
      g  = gamma[col]; be = beta[col];
      mu = mean[col];  iv = rsqrtf(var[col] + BN_EPS);
    }
    #pragma unroll
    for (int r = 0; r < 8; ++r) {
      // C/D layout: VGPR r -> M=r (lanes 0-15), M=r+8 (lanes 16-31)
      int row = rowBase + wave * 16 + half * 8 + r;
      float v = accs[j][r] + bv;
      if (MODE == 0) {
        v = fmaxf(v, 0.0f);
        outB[(size_t)row * outStride + col] = f2bf(v);
      } else {
        v = g * (v - mu) * iv + be;
        if (relu) v = fmaxf(v, 0.0f);
        outF[(size_t)row * outStride + col] = v;
      }
    }
  }
}

// ---------------------------------------------------------------------------
// segment offsets from sorted batch: offs[b] = first node of graph b; offs[B]=N
__global__ void k_offsets(const int* __restrict__ batch, int* __restrict__ offs) {
  int stride = gridDim.x * blockDim.x;
  for (int i = blockIdx.x * blockDim.x + threadIdx.x; i < N_NODES; i += stride) {
    int cur  = batch[i];
    int prev = (i == 0) ? -1 : batch[i - 1];
    for (int b = prev + 1; b <= cur; ++b) offs[b] = i;
    if (i == N_NODES - 1) offs[NGRAPH] = N_NODES;
  }
}

__global__ void k_zero(float* __restrict__ p, size_t n) {
  size_t stride = (size_t)gridDim.x * blockDim.x;
  for (size_t i = (size_t)blockIdx.x * blockDim.x + threadIdx.x; i < n; i += stride)
    p[i] = 0.0f;
}

__global__ void k_pack(const float* __restrict__ h, const int* __restrict__ batch,
                       const int* __restrict__ offs, float* __restrict__ out) {
  int i = blockIdx.x;
  int b = batch[i];
  int p = i - offs[b];
  if (p >= MAXN) return;                       // matches JAX OOB-drop semantics
  float* dst = out + ((size_t)b * MAXN + p) * DIM;
  const float* src = h + (size_t)i * DIM;
  for (int d = threadIdx.x; d < DIM; d += blockDim.x) dst[d] = src[d];
}

__global__ void k_mask(const int* __restrict__ offs, float* __restrict__ mask) {
  int b = blockIdx.x;
  int cnt = offs[b + 1] - offs[b];
  for (int t = threadIdx.x; t < MAXN; t += blockDim.x)
    mask[(size_t)b * MAXN + t] = (t >= cnt) ? 1.0f : 0.0f;
}

// ---------------------------------------------------------------------------
extern "C" void kernel_launch(void* const* d_in, const int* in_sizes, int n_in,
                              void* d_out, int out_size, void* d_ws, size_t ws_size,
                              hipStream_t stream) {
  (void)in_sizes; (void)n_in; (void)out_size; (void)ws_size;

  const int*   x          = (const int*)d_in[0];
  const int*   edge_index = (const int*)d_in[1];
  const int*   edge_attr  = (const int*)d_in[2];
  const int*   batch      = (const int*)d_in[3];
  /* d_in[4] = max_nodes (static 96) */
  const float* atom_emb1  = (const float*)d_in[5];
  const float* atom_emb2  = (const float*)d_in[6];
  const float* edge_emb1  = (const float*)d_in[7];
  const float* edge_emb2  = (const float*)d_in[8];
  const float* W1  = (const float*)d_in[9];
  const float* b1  = (const float*)d_in[10];
  const float* W2  = (const float*)d_in[11];
  const float* b2  = (const float*)d_in[12];
  const float* bnG = (const float*)d_in[13];
  const float* bnB = (const float*)d_in[14];
  const float* bnM = (const float*)d_in[15];
  const float* bnV = (const float*)d_in[16];

  // workspace carve-out (~560 MB total)
  char* ws = (char*)d_ws;
  size_t off = 0;
  auto carve = [&](size_t bytes) -> void* {
    void* p = ws + off;
    off = (off + bytes + 255) & ~(size_t)255;
    return p;
  };
  float* h    = (float*)carve((size_t)N_NODES * DIM * sizeof(float));
  float* agg  = (float*)carve((size_t)N_NODES * DIM * sizeof(float));
  bf16*  Abf  = (bf16*) carve((size_t)N_NODES * KPAD1 * sizeof(bf16));
  bf16*  Tbf  = (bf16*) carve((size_t)N_NODES * KPAD2 * sizeof(bf16));
  bf16*  W1T  = (bf16*) carve((size_t)NPAD1 * KPAD1 * sizeof(bf16));
  bf16*  W2T  = (bf16*) carve((size_t)NPAD2 * KPAD2 * sizeof(bf16));
  int*   offs = (int*)  carve((size_t)(NGRAPH + 1) * sizeof(int));

  k_init_h<<<N_NODES, 256, 0, stream>>>(x, atom_emb1, atom_emb2, h);

  for (int l = 0; l < NLAYER; ++l) {
    const float* ee1 = edge_emb1 + (size_t)l * 6 * DIM;
    const float* ee2 = edge_emb2 + (size_t)l * 3 * DIM;

    k_init_agg_self<<<N_NODES, 256, 0, stream>>>(h, ee1, ee2, agg);
    k_edge_scatter<<<N_EDGES, 256, 0, stream>>>(h, edge_index, edge_attr, ee1, ee2, agg);
    k_conv_A<<<N_NODES, 256, 0, stream>>>(agg, Abf);
    k_conv_W1T<<<NPAD1, 256, 0, stream>>>(W1 + (size_t)l * DIM * D2, W1T);
    k_conv_W2T<<<NPAD2, 256, 0, stream>>>(W2 + (size_t)l * D2 * DIM, W2T);

    dim3 g1(N_NODES / 128, NPAD1 / 64);   // 1024 x 10
    gin_gemm_wmma<0><<<g1, 256, 0, stream>>>(
        Abf, W1T, b1 + (size_t)l * D2,
        nullptr, nullptr, nullptr, nullptr,
        0, KPAD1, D2, KPAD2, nullptr, Tbf, KPAD2);

    dim3 g2(N_NODES / 128, NPAD2 / 64);   // 1024 x 5
    gin_gemm_wmma<1><<<g2, 256, 0, stream>>>(
        Tbf, W2T, b2 + (size_t)l * DIM,
        bnG + (size_t)l * DIM, bnB + (size_t)l * DIM,
        bnM + (size_t)l * DIM, bnV + (size_t)l * DIM,
        (l < NLAYER - 1) ? 1 : 0, KPAD2, DIM, DIM, h, nullptr, DIM);
  }

  // ragged -> dense pack + mask
  k_offsets<<<256, 256, 0, stream>>>(batch, offs);
  size_t padElems = (size_t)NGRAPH * MAXN * DIM;
  k_zero<<<4096, 256, 0, stream>>>((float*)d_out, padElems);
  k_pack<<<N_NODES, 256, 0, stream>>>(h, batch, offs, (float*)d_out);
  k_mask<<<NGRAPH, 128, 0, stream>>>(offs, (float*)d_out + padElems);
}
